// EV_GCN_32401233281113
// MI455X (gfx1250) — matrix operand order repaired and checked by reading, not compiled, not run on hardware
//
#include <hip/hip_runtime.h>
#include <hip/hip_bf16.h>

#define N_NODES   20000
#define E_EDGES   320000
#define D_FEAT    128
#define H_HEADS   6
#define C_CH      64
#define LG_LAYERS 4
#define NC_CLS    2
#define EHALF     64
#define PHID      128
#define AHID      16
#define HC        (H_HEADS * C_CH)   // 384

#define BN_S 0.99999500003749968f    // 1/sqrt(1+1e-5)

typedef __attribute__((ext_vector_type(16))) __bf16 v16bf;
typedef __attribute__((ext_vector_type(8)))  __bf16 v8bf;
typedef __attribute__((ext_vector_type(8)))  float  v8f;
typedef unsigned int u32x4 __attribute__((ext_vector_type(4)));
typedef int          i32x8 __attribute__((ext_vector_type(8)));
typedef int          i32x4 __attribute__((ext_vector_type(4)));

__device__ inline __bf16 f2bf(float f) { return (__bf16)f; }   // native v_cvt
__device__ inline float  bf2f(__bf16 b) { return (float)b; }

__device__ inline v8f wmma_bf16(v16bf a, v16bf b, v8f c) {
  // D = A(16x32 bf16) x B(32x16 bf16) + C(16x16 f32)
  return __builtin_amdgcn_wmma_f32_16x16x32_bf16(false, a, false, b, (short)0, c,
                                                 false, false);
}

__device__ inline void atomicMaxF(float* addr, float v) {
  if (v >= 0.f) atomicMax((int*)addr, __float_as_int(v));
  else          atomicMin((unsigned int*)addr, __float_as_uint(v));
}

// 1-D TDM copy: n8 * 8 bytes from global 'gaddr' -> LDS byte offset 'lds_addr'.
// D# per CDNA5 ISA ch.8: group0 = {flags, lds_addr, gaddr lo, gaddr hi|type=2},
// group1 = data_size=8B, tensor_dim0=n8, tile_dim0=n8 (<=0xFFFF), 1 row.
// Toolchain here exposes the 6-arg builtin (extra int32x8 group before cpol).
__device__ inline void tdm_copy_to_lds(unsigned lds_addr, const void* gaddr,
                                       unsigned n8) {
  unsigned long long ga = (unsigned long long)gaddr;
  u32x4 g0;
  g0[0] = 1u;                                            // count=1, user mode
  g0[1] = lds_addr;                                      // lds_addr
  g0[2] = (unsigned)(ga & 0xFFFFFFFFu);                  // global_addr[31:0]
  g0[3] = (unsigned)((ga >> 32) & 0x01FFFFFFu) | 0x80000000u;  // [56:32]|type=2
  i32x8 g1;
  g1[0] = (int)(3u << 16);                               // data_size = 8 bytes
  g1[1] = (int)((n8 & 0xFFFFu) << 16);                   // tensor_dim0[15:0]
  g1[2] = (int)((n8 >> 16) & 0xFFFFu) | (1 << 16);       // td0[31:16] | td1=1
  g1[3] = (int)(n8 << 16);                               // tile_dim0 = n8
  g1[4] = 1;                                             // tile_dim1 = 1
  g1[5] = (int)n8;                                       // tensor_dim0_stride
  g1[6] = 0;
  g1[7] = 0;
  i32x4 z4 = {0, 0, 0, 0};
  i32x8 z8 = {0, 0, 0, 0, 0, 0, 0, 0};
  __builtin_amdgcn_tensor_load_to_lds(g0, g1, z4, z4, z8, 0);
}

// ---------------------------------------------------------------- utilities
__global__ void k_f32_to_bf16(const float* __restrict__ src,
                              __bf16* __restrict__ dst, int n) {
  int i = blockIdx.x * blockDim.x + threadIdx.x;
  if (i < n) dst[i] = f2bf(src[i]);
}

// transpose + convert: src f32 [K, ncol] -> dst bf16 [ncol, K]
__global__ void k_cvt_t(const float* __restrict__ src, __bf16* __restrict__ dst,
                        int K, int ncol) {
  int i = blockIdx.x * blockDim.x + threadIdx.x;
  if (i >= K * ncol) return;
  int n = i / K, k = i - n * K;
  dst[i] = f2bf(src[k * ncol + n]);
}

__global__ void k_zero_f32(float* __restrict__ p, int n) {
  int i = blockIdx.x * blockDim.x + threadIdx.x;
  if (i < n) p[i] = 0.f;
}

// ---------------------------------------------------------------- PAE (fused)
// One wave processes a 16-edge tile: two shared-MLP halves, cosine similarity.
__device__ inline void pae_half(const float* __restrict__ xrow,
                                const __bf16* __restrict__ sW1t,
                                const __bf16* __restrict__ sW2t,
                                const float* __restrict__ sB1,
                                const float* __restrict__ sGS,
                                const float* __restrict__ sBE,
                                const float* __restrict__ sB2,
                                __bf16* __restrict__ hst,
                                int lhalf, int hi, v8f out[8]) {
  // ---- GEMM1: hidden = bn(relu(x @ W1 + b1)), x tile 16x64, W1 64x128
  v16bf a1[2];
#pragma unroll
  for (int ks = 0; ks < 2; ++ks) {
    int klo = ks * 32 + hi * 8;
    float4 f0 = *(const float4*)(xrow + klo);
    float4 f1 = *(const float4*)(xrow + klo + 4);
    float4 f2 = *(const float4*)(xrow + klo + 16);
    float4 f3 = *(const float4*)(xrow + klo + 20);
    v16bf a;
    a[0] = f2bf(f0.x);  a[1] = f2bf(f0.y);  a[2] = f2bf(f0.z);  a[3] = f2bf(f0.w);
    a[4] = f2bf(f1.x);  a[5] = f2bf(f1.y);  a[6] = f2bf(f1.z);  a[7] = f2bf(f1.w);
    a[8] = f2bf(f2.x);  a[9] = f2bf(f2.y);  a[10] = f2bf(f2.z); a[11] = f2bf(f2.w);
    a[12] = f2bf(f3.x); a[13] = f2bf(f3.y); a[14] = f2bf(f3.z); a[15] = f2bf(f3.w);
    a1[ks] = a;
  }
#pragma unroll
  for (int nt = 0; nt < 8; ++nt) {
    v8f c = {};
#pragma unroll
    for (int ks = 0; ks < 2; ++ks) {
      const __bf16* brow = sW1t + (nt * 16 + lhalf) * EHALF + ks * 32 + hi * 8;
      v16bf b = __builtin_shufflevector(*(const v8bf*)brow, *(const v8bf*)(brow + 16), 0, 1, 2, 3, 4, 5, 6, 7, 8, 9, 10, 11, 12, 13, 14, 15);
      c = wmma_bf16(a1[ks], b, c);
    }
    int n = nt * 16 + lhalf;
    float b1v = sB1[n], gsv = sGS[n], bev = sBE[n];
#pragma unroll
    for (int r = 0; r < 8; ++r) {
      int m = hi * 8 + r;
      float h = fmaxf(c[r] + b1v, 0.f) * gsv + bev;
      hst[m * PHID + n] = f2bf(h);
    }
  }
  // ---- GEMM2: out = hidden @ W2 + b2, hidden 16x128, W2 128x128
  v16bf a2[4];
  const __bf16* hrow = hst + lhalf * PHID;
#pragma unroll
  for (int ks = 0; ks < 4; ++ks) {
    int klo = ks * 32 + hi * 8;
    v8bf lo = *(const v8bf*)(hrow + klo);
    v8bf hh = *(const v8bf*)(hrow + klo + 16);
    a2[ks] = __builtin_shufflevector(lo, hh, 0, 1, 2, 3, 4, 5, 6, 7, 8, 9, 10,
                                     11, 12, 13, 14, 15);
  }
#pragma unroll
  for (int nt = 0; nt < 8; ++nt) {
    v8f c = {};
#pragma unroll
    for (int ks = 0; ks < 4; ++ks) {
      const __bf16* brow = sW2t + (nt * 16 + lhalf) * PHID + ks * 32 + hi * 8;
      v16bf b = __builtin_shufflevector(*(const v8bf*)brow, *(const v8bf*)(brow + 16), 0, 1, 2, 3, 4, 5, 6, 7, 8, 9, 10, 11, 12, 13, 14, 15);
      c = wmma_bf16(a2[ks], b, c);
    }
    float b2v = sB2[nt * 16 + lhalf];
#pragma unroll
    for (int r = 0; r < 8; ++r) c[r] += b2v;
    out[nt] = c;
  }
}

__global__ __launch_bounds__(64) void k_pae(const float* __restrict__ xin,
                                            const __bf16* __restrict__ Wt,
                                            const float* __restrict__ B1,
                                            const float* __restrict__ G1,
                                            const float* __restrict__ BE1,
                                            const float* __restrict__ B2,
                                            float* __restrict__ ew) {
  // Wt = pre-transposed bf16 weights: [128][64] W1t then [128][128] W2t
  __shared__ __bf16 sW[PHID * EHALF + PHID * PHID];            // 48KB
  __shared__ float  sB1[PHID], sGS[PHID], sBE[PHID], sB2[PHID];  // 2KB
  __shared__ __bf16 sHid[2][16 * PHID];                        // 8KB

  const int tid = threadIdx.x;
  // TDM: one DMA of the whole transposed weight block into LDS.
  if (tid < 32) {
    unsigned ldsW = (unsigned)(unsigned long long)(void*)sW;
    tdm_copy_to_lds(ldsW, Wt, (PHID * EHALF + PHID * PHID) * 2 / 8);
  }
  for (int i = tid; i < PHID; i += 64) {
    sB1[i] = B1[i];
    sGS[i] = G1[i] * BN_S;
    sBE[i] = BE1[i];
    sB2[i] = B2[i];
  }
  __builtin_amdgcn_s_wait_tensorcnt(0);
  __syncthreads();

  const __bf16* sW1t = sW;
  const __bf16* sW2t = sW + PHID * EHALF;

  const int wid = tid >> 5;
  const int lane = tid & 31;
  const int lhalf = lane & 15;
  const int hi = (lane >> 4) & 1;
  const int e0 = (blockIdx.x * 2 + wid) * 16;
  __bf16* hst = &sHid[wid][0];

  v8f o1[8], o2[8];
  const float* xr = xin + (size_t)(e0 + lhalf) * (2 * EHALF);
  pae_half(xr,         sW1t, sW2t, sB1, sGS, sBE, sB2, hst, lhalf, hi, o1);
  pae_half(xr + EHALF, sW1t, sW2t, sB1, sGS, sBE, sB2, hst, lhalf, hi, o2);

  float pd[8], p1s[8], p2s[8];
#pragma unroll
  for (int r = 0; r < 8; ++r) { pd[r] = 0.f; p1s[r] = 0.f; p2s[r] = 0.f; }
#pragma unroll
  for (int nt = 0; nt < 8; ++nt) {
#pragma unroll
    for (int r = 0; r < 8; ++r) {
      float x1 = o1[nt][r], x2 = o2[nt][r];
      pd[r] += x1 * x2;
      p1s[r] += x1 * x1;
      p2s[r] += x2 * x2;
    }
  }
  // reduce over the 16 lanes of each half-group (masks < 16 stay in-group)
#pragma unroll
  for (int r = 0; r < 8; ++r) {
#pragma unroll
    for (int mask = 1; mask < 16; mask <<= 1) {
      pd[r]  += __shfl_xor(pd[r], mask);
      p1s[r] += __shfl_xor(p1s[r], mask);
      p2s[r] += __shfl_xor(p2s[r], mask);
    }
  }
  if (lhalf == 0) {
#pragma unroll
    for (int r = 0; r < 8; ++r) {
      float cosv = pd[r] / (sqrtf(p1s[r]) * sqrtf(p2s[r]) + 1e-8f);
      ew[e0 + hi * 8 + r] = (cosv + 1.f) * 0.5f;
    }
  }
}

// ---------------------------------------------------------------- generic GEMM
// C[M, ncol] = A[M, K] @ W[K, ncol] ; K = NKS*32 compile-time.
// Wt = pre-transposed bf16 weights [ncol][K] (tile rows are contiguous -> TDM).
// MODE 0: store f32 raw. MODE 1: bn(relu(x+bias)) -> bf16.
template <int NKS, int MODE>
__global__ __launch_bounds__(128) void k_gemm(const __bf16* __restrict__ A, int M,
                                              const __bf16* __restrict__ Wt,
                                              float* __restrict__ outF,
                                              __bf16* __restrict__ outBF, int ldo,
                                              const float* __restrict__ bias,
                                              const float* __restrict__ gsc,
                                              const float* __restrict__ bsh) {
  constexpr int K = NKS * 32;
  __shared__ __bf16 sWt[64 * 128];  // [n=64][k<=128]
  const int tid = threadIdx.x;
  const int n0 = blockIdx.y * 64;
  if (tid < 32) {
    unsigned ldsW = (unsigned)(unsigned long long)(void*)sWt;
    tdm_copy_to_lds(ldsW, Wt + (size_t)n0 * K, 64 * K * 2 / 8);
  }
  __builtin_amdgcn_s_wait_tensorcnt(0);
  __syncthreads();

  const int wid = tid >> 5, lane = tid & 31;
  const int lhalf = lane & 15, hi = (lane >> 4) & 1;
  const int m0 = blockIdx.x * 64 + wid * 16;
  int mrow = m0 + lhalf;
  if (mrow >= M) mrow = M - 1;
  const __bf16* arow = A + (size_t)mrow * K;

  v16bf a[NKS];
#pragma unroll
  for (int ks = 0; ks < NKS; ++ks) {
    int klo = ks * 32 + hi * 8;
    v8bf lo = *(const v8bf*)(arow + klo);
    v8bf hh = *(const v8bf*)(arow + klo + 16);
    a[ks] = __builtin_shufflevector(lo, hh, 0, 1, 2, 3, 4, 5, 6, 7, 8, 9, 10, 11,
                                    12, 13, 14, 15);
  }
#pragma unroll
  for (int nt = 0; nt < 4; ++nt) {
    v8f c = {};
#pragma unroll
    for (int ks = 0; ks < NKS; ++ks) {
      const __bf16* brow = sWt + (nt * 16 + lhalf) * K + ks * 32 + hi * 8;
      v16bf b = __builtin_shufflevector(*(const v8bf*)brow, *(const v8bf*)(brow + 16), 0, 1, 2, 3, 4, 5, 6, 7, 8, 9, 10, 11, 12, 13, 14, 15);
      c = wmma_bf16(a[ks], b, c);
    }
    int n = n0 + nt * 16 + lhalf;
#pragma unroll
    for (int r = 0; r < 8; ++r) {
      int m = m0 + hi * 8 + r;
      if (m < M) {
        if (MODE == 1) {
          float v = fmaxf(c[r] + bias[n], 0.f) * gsc[n] * BN_S + bsh[n];
          outBF[(size_t)m * ldo + n] = f2bf(v);
        } else {
          outF[(size_t)m * ldo + n] = c[r];
        }
      }
    }
  }
}

// ---------------------------------------------------------------- GAT pieces
__global__ void k_alpha_init(const float* __restrict__ xw,
                             const float* __restrict__ asrc,
                             const float* __restrict__ adst,
                             float* __restrict__ als, float* __restrict__ ald,
                             float* __restrict__ mmax, float* __restrict__ den) {
  int i = blockIdx.x * blockDim.x + threadIdx.x;
  if (i >= N_NODES * H_HEADS) return;
  int node = i / H_HEADS, h = i - node * H_HEADS;
  const float* row = xw + (size_t)node * HC + h * C_CH;
  float s = 0.f, d = 0.f;
  for (int c = 0; c < C_CH; ++c) {
    float v = row[c];
    s += v * asrc[h * C_CH + c];
    d += v * adst[h * C_CH + c];
  }
  als[i] = s;
  ald[i] = d;
  mmax[i] = -1e30f;
  den[i] = 0.f;
}

__global__ void k_edge_logit(const int* __restrict__ ei,
                             const float* __restrict__ als,
                             const float* __restrict__ ald,
                             float* __restrict__ ebuf, float* __restrict__ mmax) {
  int i = blockIdx.x * blockDim.x + threadIdx.x;
  if (i >= E_EDGES * H_HEADS) return;
  int e = i / H_HEADS, h = i - e * H_HEADS;
  int src = ei[e], dst = ei[E_EDGES + e];
  float v = als[src * H_HEADS + h] + ald[dst * H_HEADS + h];
  v = (v >= 0.f) ? v : 0.3f * v;  // leaky relu 0.3
  ebuf[i] = v;
  atomicMaxF(&mmax[dst * H_HEADS + h], v);
}

__global__ void k_edge_exp(const int* __restrict__ ei,
                           float* __restrict__ ebuf,
                           const float* __restrict__ mmax,
                           float* __restrict__ den) {
  int i = blockIdx.x * blockDim.x + threadIdx.x;
  if (i >= E_EDGES * H_HEADS) return;
  int e = i / H_HEADS, h = i - e * H_HEADS;
  int dst = ei[E_EDGES + e];
  float ex = __expf(ebuf[i] - mmax[dst * H_HEADS + h]);
  ebuf[i] = ex;
  atomicAdd(&den[dst * H_HEADS + h], ex);
}

__global__ void k_edge_msg(const int* __restrict__ ei,
                           const float* __restrict__ ebuf,
                           const float* __restrict__ den,
                           const float* __restrict__ ew,
                           const float* __restrict__ xw,
                           float* __restrict__ aggr) {
  int i = blockIdx.x * blockDim.x + threadIdx.x;
  if (i >= E_EDGES * H_HEADS) return;
  int e = i / H_HEADS, h = i - e * H_HEADS;
  int src = ei[e], dst = ei[E_EDGES + e];
  float w = ebuf[i] / (den[dst * H_HEADS + h] + 1e-16f) * ew[e];
  const float* xr = xw + (size_t)src * HC + h * C_CH;
  float* ar = aggr + (size_t)dst * HC + h * C_CH;
  for (int c = 0; c < C_CH; c += 4) {
    float4 v = *(const float4*)(xr + c);
    atomicAdd(&ar[c + 0], v.x * w);
    atomicAdd(&ar[c + 1], v.y * w);
    atomicAdd(&ar[c + 2], v.z * w);
    atomicAdd(&ar[c + 3], v.w * w);
  }
}

__global__ void k_node_out(const float* __restrict__ aggr,
                           float* __restrict__ embs,
                           __bf16* __restrict__ hbf, int layer) {
  int i = blockIdx.x * blockDim.x + threadIdx.x;
  if (i >= N_NODES * C_CH) return;
  int node = i >> 6, c = i & 63;
  float s = 0.f;
  for (int h = 0; h < H_HEADS; ++h) s += aggr[(size_t)node * HC + h * C_CH + c];
  s = fmaxf(s * (1.f / H_HEADS), 0.f);
  embs[(size_t)node * (LG_LAYERS * C_CH) + layer * C_CH + c] = s;
  hbf[(size_t)node * C_CH + c] = f2bf(s);
}

// ---------------------------------------------------------------- attention pool
__global__ void k_att_pool(const float* __restrict__ embs,
                           const float* __restrict__ W1,
                           const float* __restrict__ B1,
                           const float* __restrict__ W2,
                           __bf16* __restrict__ embbf) {
  int n = blockIdx.x * blockDim.x + threadIdx.x;
  if (n >= N_NODES) return;
  const float* zr = embs + (size_t)n * (LG_LAYERS * C_CH);
  float wl[LG_LAYERS];
#pragma unroll
  for (int l = 0; l < LG_LAYERS; ++l) {
    float acc[AHID];
#pragma unroll
    for (int j = 0; j < AHID; ++j) acc[j] = B1[j];
    for (int c = 0; c < C_CH; ++c) {
      float zv = zr[l * C_CH + c];
#pragma unroll
      for (int j = 0; j < AHID; ++j) acc[j] += zv * W1[c * AHID + j];
    }
    float s = 0.f;
#pragma unroll
    for (int j = 0; j < AHID; ++j) s += tanhf(acc[j]) * W2[j];
    wl[l] = s;
  }
  float m = wl[0];
#pragma unroll
  for (int l = 1; l < LG_LAYERS; ++l) m = fmaxf(m, wl[l]);
  float ssum = 0.f;
#pragma unroll
  for (int l = 0; l < LG_LAYERS; ++l) { wl[l] = __expf(wl[l] - m); ssum += wl[l]; }
  float inv = 1.f / ssum;
  for (int c = 0; c < C_CH; ++c) {
    float s = 0.f;
#pragma unroll
    for (int l = 0; l < LG_LAYERS; ++l) s += wl[l] * inv * zr[l * C_CH + c];
    embbf[(size_t)n * C_CH + c] = f2bf(s);
  }
}

// ---------------------------------------------------------------- classifier out
__global__ void k_cls_out(const __bf16* __restrict__ hcl,
                          const float* __restrict__ W2,
                          const float* __restrict__ B2,
                          float* __restrict__ logits) {
  int n = blockIdx.x * blockDim.x + threadIdx.x;
  if (n >= N_NODES) return;
  float a0 = B2[0], a1 = B2[1];
  const __bf16* hr = hcl + (size_t)n * 256;
  for (int k = 0; k < 256; ++k) {
    float hv = bf2f(hr[k]);
    a0 += hv * W2[k * NC_CLS + 0];
    a1 += hv * W2[k * NC_CLS + 1];
  }
  logits[n * NC_CLS + 0] = a0;
  logits[n * NC_CLS + 1] = a1;
}

// ---------------------------------------------------------------- launcher
extern "C" void kernel_launch(void* const* d_in, const int* in_sizes, int n_in,
                              void* d_out, int out_size, void* d_ws, size_t ws_size,
                              hipStream_t stream) {
  (void)in_sizes; (void)out_size; (void)ws_size;
  const float* features = (const float*)d_in[0];
  const int*   eidx     = (const int*)d_in[1];
  const float* edgein   = (const float*)d_in[2];
  const float* paeW1 = (const float*)d_in[3];
  const float* paeB1 = (const float*)d_in[4];
  const float* paeG1 = (const float*)d_in[5];
  const float* paeBE1 = (const float*)d_in[6];
  const float* paeW2 = (const float*)d_in[7];
  const float* paeB2 = (const float*)d_in[8];

  const float *gatW[LG_LAYERS], *gasrc[LG_LAYERS], *gadst[LG_LAYERS];
  int p = 9;
  if (n_in >= 30) {  // tuples expanded into separate inputs
    for (int l = 0; l < LG_LAYERS; ++l) gatW[l] = (const float*)d_in[p++];
    for (int l = 0; l < LG_LAYERS; ++l) gasrc[l] = (const float*)d_in[p++];
    for (int l = 0; l < LG_LAYERS; ++l) gadst[l] = (const float*)d_in[p++];
  } else {           // tuples flattened/concatenated
    const float* wAll = (const float*)d_in[p++];
    gatW[0] = wAll;
    gatW[1] = wAll + D_FEAT * HC;
    gatW[2] = gatW[1] + C_CH * HC;
    gatW[3] = gatW[2] + C_CH * HC;
    const float* sAll = (const float*)d_in[p++];
    const float* dAll = (const float*)d_in[p++];
    for (int l = 0; l < LG_LAYERS; ++l) {
      gasrc[l] = sAll + l * HC;
      gadst[l] = dAll + l * HC;
    }
  }
  const float* attW1 = (const float*)d_in[p++];
  const float* attB1 = (const float*)d_in[p++];
  const float* attW2 = (const float*)d_in[p++];
  const float* clsW1 = (const float*)d_in[p++];
  const float* clsB1 = (const float*)d_in[p++];
  const float* clsG  = (const float*)d_in[p++];
  const float* clsBE = (const float*)d_in[p++];
  const float* clsW2 = (const float*)d_in[p++];
  const float* clsB2 = (const float*)d_in[p++];

  float* logits = (float*)d_out;
  float* ew = logits + N_NODES * NC_CLS;

  char* base = (char*)d_ws;
  size_t off = 0;
  auto alloc = [&](size_t bytes) -> char* {
    char* r = base + off;
    off = (off + bytes + 255) & ~(size_t)255;
    return r;
  };
  float*  xw    = (float*)alloc(sizeof(float) * N_NODES * HC);
  float*  aggr  = (float*)alloc(sizeof(float) * N_NODES * HC);
  float*  als   = (float*)alloc(sizeof(float) * N_NODES * H_HEADS);
  float*  ald   = (float*)alloc(sizeof(float) * N_NODES * H_HEADS);
  float*  mmax  = (float*)alloc(sizeof(float) * N_NODES * H_HEADS);
  float*  den   = (float*)alloc(sizeof(float) * N_NODES * H_HEADS);
  float*  ebuf  = (float*)alloc(sizeof(float) * E_EDGES * H_HEADS);
  float*  embs  = (float*)alloc(sizeof(float) * N_NODES * LG_LAYERS * C_CH);
  __bf16* featbf = (__bf16*)alloc(2 * (size_t)N_NODES * D_FEAT);
  __bf16* hbf    = (__bf16*)alloc(2 * (size_t)N_NODES * C_CH);
  __bf16* wpae   = (__bf16*)alloc(2 * (size_t)(PHID * EHALF + PHID * PHID));
  __bf16* wbfT   = (__bf16*)alloc(2 * (size_t)HC * D_FEAT);
  __bf16* embbf  = (__bf16*)alloc(2 * (size_t)N_NODES * C_CH);
  __bf16* hclbf  = (__bf16*)alloc(2 * (size_t)N_NODES * 256);

  const int TB = 256;
  // features -> bf16 (row-major)
  k_f32_to_bf16<<<(N_NODES * D_FEAT + TB - 1) / TB, TB, 0, stream>>>(
      features, featbf, N_NODES * D_FEAT);
  // PAE weights -> transposed bf16 (contiguous W1t then W2t), then fused PAE
  k_cvt_t<<<(PHID * EHALF + TB - 1) / TB, TB, 0, stream>>>(paeW1, wpae, EHALF,
                                                           PHID);
  k_cvt_t<<<(PHID * PHID + TB - 1) / TB, TB, 0, stream>>>(
      paeW2, wpae + PHID * EHALF, PHID, PHID);
  k_pae<<<E_EDGES / 32, 64, 0, stream>>>(edgein, wpae, paeB1, paeG1, paeBE1,
                                         paeB2, ew);

  for (int l = 0; l < LG_LAYERS; ++l) {
    const int K = (l == 0) ? D_FEAT : C_CH;
    const __bf16* Aptr = (l == 0) ? featbf : hbf;
    k_cvt_t<<<(K * HC + TB - 1) / TB, TB, 0, stream>>>(gatW[l], wbfT, K, HC);
    dim3 gg((N_NODES + 63) / 64, HC / 64);
    if (l == 0)
      k_gemm<4, 0><<<gg, 128, 0, stream>>>(Aptr, N_NODES, wbfT, xw, nullptr, HC,
                                           nullptr, nullptr, nullptr);
    else
      k_gemm<2, 0><<<gg, 128, 0, stream>>>(Aptr, N_NODES, wbfT, xw, nullptr, HC,
                                           nullptr, nullptr, nullptr);
    k_alpha_init<<<(N_NODES * H_HEADS + TB - 1) / TB, TB, 0, stream>>>(
        xw, gasrc[l], gadst[l], als, ald, mmax, den);
    k_zero_f32<<<(N_NODES * HC + TB - 1) / TB, TB, 0, stream>>>(aggr,
                                                                N_NODES * HC);
    k_edge_logit<<<(E_EDGES * H_HEADS + TB - 1) / TB, TB, 0, stream>>>(
        eidx, als, ald, ebuf, mmax);
    k_edge_exp<<<(E_EDGES * H_HEADS + TB - 1) / TB, TB, 0, stream>>>(
        eidx, ebuf, mmax, den);
    k_edge_msg<<<(E_EDGES * H_HEADS + TB - 1) / TB, TB, 0, stream>>>(
        eidx, ebuf, den, ew, xw, aggr);
    k_node_out<<<(N_NODES * C_CH + TB - 1) / TB, TB, 0, stream>>>(aggr, embs,
                                                                  hbf, l);
  }

  k_att_pool<<<(N_NODES + 127) / 128, 128, 0, stream>>>(embs, attW1, attB1,
                                                        attW2, embbf);
  // classifier: W1 [64,256] -> transposed bf16 [256][64]
  k_cvt_t<<<(C_CH * 256 + TB - 1) / TB, TB, 0, stream>>>(clsW1, wbfT, C_CH, 256);
  k_gemm<2, 1><<<dim3((N_NODES + 63) / 64, 256 / 64), 128, 0, stream>>>(
      embbf, N_NODES, wbfT, nullptr, hclbf, 256, clsB1, clsG, clsBE);
  k_cls_out<<<(N_NODES + TB - 1) / TB, TB, 0, stream>>>(hclbf, clsW2, clsB2,
                                                        logits);
}
